// AttentionEncoder_77927886618806
// MI455X (gfx1250) — matrix operand order repaired and checked by reading, not compiled, not run on hardware
//
#include <hip/hip_runtime.h>
#include <hip/hip_bf16.h>

#define B_SZ 256
#define T_SZ 512
#define D_SZ 64
#define H_SZ 128
#define G_SZ 512  // 4*H
#define BT   2    // batch rows per workgroup (scan kernel)

typedef __attribute__((ext_vector_type(16))) __bf16 v16bf;
typedef __attribute__((ext_vector_type(8)))  float  v8f;

union Frag16 { v16bf v; uint4 q[2]; unsigned short h[16]; };

static __device__ __forceinline__ unsigned short f2bf(float x) {
    unsigned int u = __float_as_uint(x);
    unsigned int r = (u + 0x7FFFu + ((u >> 16) & 1u)) >> 16;   // RNE
    return (unsigned short)r;
}
static __device__ __forceinline__ float bf2f(unsigned short h) {
    return __uint_as_float(((unsigned int)h) << 16);
}

// A fragment (16x32 bf16, M x K): lane<16: K0..15 of row lane%16;
// lane>=16: K16..31.
static __device__ __forceinline__ v16bf load_fragA(const unsigned short* base, int stride, int lane) {
    int r  = lane & 15;
    int hi = lane >> 4;
    const unsigned short* p = base + r * stride + hi * 16;
    Frag16 f;
    f.q[0] = *(const uint4*)(p);
    f.q[1] = *(const uint4*)(p + 8);
    return f.v;
}
// B fragment (32x16 bf16, K x N) with B[k][n] = W[n][k], row-major W (rows contiguous in k):
// lane<16: K0..15 of column lane%16; lane>=16: K16..31.
static __device__ __forceinline__ v16bf load_fragB(const unsigned short* base, int stride, int lane) {
    int r  = lane & 15;
    int hi = lane >> 4;
    const unsigned short* p = base + r * stride + hi * 16;
    Frag16 f;
    f.q[0] = *(const uint4*)(p);
    f.q[1] = *(const uint4*)(p + 8);
    return f.v;
}

static __device__ __forceinline__ v8f wmma_bf16(v16bf a, v16bf b, v8f c) {
    return __builtin_amdgcn_wmma_f32_16x16x32_bf16(false, a, false, b, (short)0, c, false, false);
}

static __device__ __forceinline__ float fast_tanh(float x) {
#if __has_builtin(__builtin_amdgcn_tanhf)
    return __builtin_amdgcn_tanhf(x);
#else
    x = fminf(fmaxf(x, -9.0f), 9.0f);
    float e = __builtin_amdgcn_exp2f(x * 2.8853900817779268f); // 2*log2(e)
    return (e - 1.0f) * __builtin_amdgcn_rcpf(e + 1.0f);
#endif
}
static __device__ __forceinline__ float fast_sig(float x) {
    return 0.5f + 0.5f * fast_tanh(0.5f * x);
}

// ---------------- weight f32 -> bf16 conversion ----------------
__global__ void cvt_kernel(const float* __restrict__ W1, const float* __restrict__ Wih,
                           const float* __restrict__ Whh,
                           unsigned short* __restrict__ W1b, unsigned short* __restrict__ Wihb,
                           unsigned short* __restrict__ Whhb) {
    int idx = blockIdx.x * blockDim.x + threadIdx.x;
    if (idx < T_SZ * 2 * H_SZ) W1b[idx]  = f2bf(W1[idx]);    // 131072
    if (idx < G_SZ * D_SZ)     Wihb[idx] = f2bf(Wih[idx]);   // 32768
    if (idx < G_SZ * H_SZ)     Whhb[idx] = f2bf(Whh[idx]);   // 65536
}

// ---------------- s2 precompute: s2[b][d][s] = sum_t x[b][t][d]*W2[s][t] + b2[s] ----------------
__global__ __launch_bounds__(1024) void s2_kernel(const float* __restrict__ drv,
                                                  const float* __restrict__ W2,
                                                  const float* __restrict__ b2,
                                                  unsigned short* __restrict__ s2b) {
    __shared__ __align__(16) unsigned short xT[D_SZ * 32]; // transposed x chunk: [d][t_local]
    int b    = blockIdx.x;
    int tid  = threadIdx.x;
    int lane = tid & 31;
    int wave = tid >> 5;          // N-tile (over s), 32 tiles of 16
    const float* xb = drv + (size_t)b * T_SZ * D_SZ;

    v8f acc[4];
#pragma unroll
    for (int mt = 0; mt < 4; ++mt)
#pragma unroll
        for (int i = 0; i < 8; ++i) acc[mt][i] = 0.0f;

    for (int kc = 0; kc < 16; ++kc) {  // K = t, chunks of 32
        __syncthreads();
#pragma unroll
        for (int p = 0; p < 2; ++p) {  // stage 32x64 f32 -> LDS transposed bf16
            int i  = tid + p * 1024;
            int tl = i >> 6, d = i & 63;
            xT[d * 32 + tl] = f2bf(xb[(size_t)(kc * 32 + tl) * D_SZ + d]);
        }
        __syncthreads();

        int n  = wave * 16 + (lane & 15);
        int hi = lane >> 4;
        const float4* wq = (const float4*)(W2 + (size_t)n * T_SZ + kc * 32 + hi * 16);
        float tf[16];
        *(float4*)(tf + 0)  = wq[0];
        *(float4*)(tf + 4)  = wq[1];
        *(float4*)(tf + 8)  = wq[2];
        *(float4*)(tf + 12) = wq[3];
        Frag16 fb;
#pragma unroll
        for (int j = 0; j < 16; ++j) fb.h[j] = f2bf(tf[j]);

#pragma unroll
        for (int mt = 0; mt < 4; ++mt) {
            v16bf fa = load_fragA(xT + mt * 16 * 32, 32, lane);
            acc[mt] = wmma_bf16(fa, fb.v, acc[mt]);
        }
    }

    int n  = wave * 16 + (lane & 15);
    int hi = lane >> 4;
    float bias = b2[n];
#pragma unroll
    for (int mt = 0; mt < 4; ++mt)
#pragma unroll
        for (int v = 0; v < 8; ++v) {
            int d = mt * 16 + v + 8 * hi;
            s2b[((size_t)b * D_SZ + d) * T_SZ + n] = f2bf(acc[mt][v] + bias);
        }
}

// ---------------- fused recurrent scan: one workgroup owns BT batch rows ----------------
// s2 slice for the WG's rows lives in LDS for the whole t-loop (no global re-reads).
__global__ __launch_bounds__(1024) void scan_kernel(const float* __restrict__ drv,
                                                    const unsigned short* __restrict__ Wihb,
                                                    const unsigned short* __restrict__ Whhb,
                                                    const float* __restrict__ bih,
                                                    const float* __restrict__ bhh,
                                                    const unsigned short* __restrict__ W1b,
                                                    const float* __restrict__ b1,
                                                    const float* __restrict__ W3,
                                                    const float* __restrict__ b3,
                                                    const unsigned short* __restrict__ s2b,
                                                    float* __restrict__ out) {
    __shared__ __align__(16) unsigned short s2s[BT * D_SZ * T_SZ]; // 128 KB bf16 slice
    __shared__ __align__(16) unsigned short h_bf[16 * H_SZ];
    __shared__ __align__(16) unsigned short c_bf[16 * H_SZ];
    __shared__ __align__(16) unsigned short xs_bf[16 * D_SZ];
    __shared__ __align__(16) float c_f[BT * H_SZ];
    __shared__ __align__(16) float s1s[BT * T_SZ];
    __shared__ __align__(16) float zbuf[BT * G_SZ];
    __shared__ __align__(16) float ebuf[BT * D_SZ];
    __shared__ __align__(16) float abuf[BT * D_SZ];
    __shared__ __align__(16) float pbuf[1024];
    __shared__ __align__(16) float w3s[T_SZ];

    int tid  = threadIdx.x;
    int lane = tid & 31;
    int wave = tid >> 5;
    int bt   = blockIdx.x;          // batch tile: rows bt*BT .. bt*BT+BT-1
    float b3v = b3[0];

    // ---- one-time init: state zeros, w3, and the LDS-resident s2 slice ----
#pragma unroll
    for (int p = 0; p < 2; ++p) {
        int i = tid + p * 1024;
        h_bf[i] = 0; c_bf[i] = 0;
    }
    if (tid < 16 * D_SZ) xs_bf[tid] = 0;
    if (tid < BT * H_SZ) c_f[tid] = 0.0f;
    if (tid < T_SZ) w3s[tid] = W3[tid];
    {
        const uint4* src = (const uint4*)(s2b + (size_t)(bt * BT) * D_SZ * T_SZ);
        uint4* dst = (uint4*)s2s;
        for (int i = tid; i < (BT * D_SZ * T_SZ) / 8; i += 1024) dst[i] = src[i];
    }
    __syncthreads();

    for (int t = 0; t < T_SZ; ++t) {
        // ---- Stage A: s1 = h @ W1h^T + c @ W1c^T + b1  (wave = N-tile over T) ----
        {
            v8f acc;
#pragma unroll
            for (int i = 0; i < 8; ++i) acc[i] = 0.0f;
#pragma unroll
            for (int kc = 0; kc < 4; ++kc) {
                v16bf ah = load_fragA(h_bf + kc * 32, H_SZ, lane);
                v16bf bh = load_fragB(W1b + (size_t)(wave * 16) * 256 + kc * 32, 256, lane);
                acc = wmma_bf16(ah, bh, acc);
                v16bf ac = load_fragA(c_bf + kc * 32, H_SZ, lane);
                v16bf bc = load_fragB(W1b + (size_t)(wave * 16) * 256 + H_SZ + kc * 32, 256, lane);
                acc = wmma_bf16(ac, bc, acc);
            }
            int n  = wave * 16 + (lane & 15);
            int hi = lane >> 4;
            float bb = b1[n];
            if (hi == 0) {
#pragma unroll
                for (int v = 0; v < BT; ++v) s1s[v * T_SZ + n] = acc[v] + bb;
            }
        }
        __syncthreads();

        // ---- Stage B: e[m][d] = sum_s tanh(s1[m][s]+s2[m,d,s])*w3[s]+b3; softmax_d; x = a*x_t
        // 8 threads per (m,d) pair, each reduces 64 s-values from the LDS s2 slice.
        {
            int pair = tid >> 3;           // 0 .. BT*64-1
            int q    = tid & 7;
            int m = pair >> 6, d = pair & 63;
            const unsigned short* s2p = s2s + (size_t)pair * T_SZ + q * 64;
            const float* s1r = s1s + m * T_SZ + q * 64;
            const float* w3r = w3s + q * 64;
            float acc = 0.0f;
#pragma unroll
            for (int s0 = 0; s0 < 64; s0 += 8) {
                union { uint4 q4; unsigned short h[8]; } u;
                u.q4 = *(const uint4*)(s2p + s0);
                float4 f0 = *(const float4*)(s1r + s0);
                float4 f1 = *(const float4*)(s1r + s0 + 4);
                acc = fmaf(fast_tanh(f0.x + bf2f(u.h[0])), w3r[s0 + 0], acc);
                acc = fmaf(fast_tanh(f0.y + bf2f(u.h[1])), w3r[s0 + 1], acc);
                acc = fmaf(fast_tanh(f0.z + bf2f(u.h[2])), w3r[s0 + 2], acc);
                acc = fmaf(fast_tanh(f0.w + bf2f(u.h[3])), w3r[s0 + 3], acc);
                acc = fmaf(fast_tanh(f1.x + bf2f(u.h[4])), w3r[s0 + 4], acc);
                acc = fmaf(fast_tanh(f1.y + bf2f(u.h[5])), w3r[s0 + 5], acc);
                acc = fmaf(fast_tanh(f1.z + bf2f(u.h[6])), w3r[s0 + 6], acc);
                acc = fmaf(fast_tanh(f1.w + bf2f(u.h[7])), w3r[s0 + 7], acc);
            }
            pbuf[tid] = acc;
            __syncthreads();
            float e = b3v;
#pragma unroll
            for (int j = 0; j < 8; ++j) e += pbuf[pair * 8 + j];
            ebuf[pair] = e;                 // 8 duplicate identical writes: benign
            __syncthreads();
            float mx = -3.4e38f;
            const float* er = ebuf + m * 64;
#pragma unroll 8
            for (int j = 0; j < 64; ++j) mx = fmaxf(mx, er[j]);
            float au = __builtin_amdgcn_exp2f((e - mx) * 1.44269504f);
            abuf[pair] = au;
            __syncthreads();
            float ssum = 0.0f;
            const float* ar = abuf + m * 64;
#pragma unroll 8
            for (int j = 0; j < 64; ++j) ssum += ar[j];
            float xt = drv[((size_t)(bt * BT + m) * T_SZ + t) * D_SZ + d];
            xs_bf[pair] = f2bf(au * __builtin_amdgcn_rcpf(ssum) * xt);
        }
        __syncthreads();

        // ---- Stage C: z = x @ Wih^T + h @ Whh^T + bih + bhh  (wave = N-tile over 4H) ----
        {
            v8f acc;
#pragma unroll
            for (int i = 0; i < 8; ++i) acc[i] = 0.0f;
#pragma unroll
            for (int kc = 0; kc < 2; ++kc) {
                v16bf ax = load_fragA(xs_bf + kc * 32, D_SZ, lane);
                v16bf bw = load_fragB(Wihb + (size_t)(wave * 16) * D_SZ + kc * 32, D_SZ, lane);
                acc = wmma_bf16(ax, bw, acc);
            }
#pragma unroll
            for (int kc = 0; kc < 4; ++kc) {
                v16bf ah = load_fragA(h_bf + kc * 32, H_SZ, lane);
                v16bf bw = load_fragB(Whhb + (size_t)(wave * 16) * H_SZ + kc * 32, H_SZ, lane);
                acc = wmma_bf16(ah, bw, acc);
            }
            int n  = wave * 16 + (lane & 15);
            int hi = lane >> 4;
            float bb = bih[n] + bhh[n];
            if (hi == 0) {
#pragma unroll
                for (int v = 0; v < BT; ++v) zbuf[v * G_SZ + n] = acc[v] + bb;
            }
        }
        __syncthreads();

        // ---- Cell update + output store (BT*128 elements) ----
        if (tid < BT * H_SZ) {
            int m = tid >> 7, j = tid & 127;
            float iv = zbuf[m * G_SZ + j];
            float fv = zbuf[m * G_SZ + 128 + j];
            float gv = zbuf[m * G_SZ + 256 + j];
            float ov = zbuf[m * G_SZ + 384 + j];
            float cn = fast_sig(fv) * c_f[tid] + fast_sig(iv) * fast_tanh(gv);
            float hn = fast_sig(ov) * fast_tanh(cn);
            c_f[tid]  = cn;
            c_bf[m * H_SZ + j] = f2bf(cn);
            h_bf[m * H_SZ + j] = f2bf(hn);
            out[((size_t)(bt * BT + m)) * T_SZ * H_SZ + (size_t)t * H_SZ + j] = hn;
        }
        __syncthreads();
    }
}

extern "C" void kernel_launch(void* const* d_in, const int* in_sizes, int n_in,
                              void* d_out, int out_size, void* d_ws, size_t ws_size,
                              hipStream_t stream) {
    const float* drv = (const float*)d_in[0];
    const float* Wih = (const float*)d_in[1];
    const float* Whh = (const float*)d_in[2];
    const float* bih = (const float*)d_in[3];
    const float* bhh = (const float*)d_in[4];
    const float* W1  = (const float*)d_in[5];
    const float* b1  = (const float*)d_in[6];
    const float* W2  = (const float*)d_in[7];
    const float* b2  = (const float*)d_in[8];
    const float* W3  = (const float*)d_in[9];
    const float* b3  = (const float*)d_in[10];
    float* out = (float*)d_out;

    char* ws = (char*)d_ws;
    unsigned short* s2b  = (unsigned short*)(ws);                         // 16.78 MB
    unsigned short* W1b  = (unsigned short*)(ws + (size_t)16777216);
    unsigned short* Wihb = (unsigned short*)(ws + (size_t)16777216 + 262144);
    unsigned short* Whhb = (unsigned short*)(ws + (size_t)16777216 + 262144 + 65536);

    cvt_kernel<<<512, 256, 0, stream>>>(W1, Wih, Whh, W1b, Wihb, Whhb);
    s2_kernel<<<B_SZ, 1024, 0, stream>>>(drv, W2, b2, s2b);
    scan_kernel<<<B_SZ / BT, 1024, 0, stream>>>(drv, Wihb, Whhb, bih, bhh, W1b, b1, W3, b3, s2b, out);
}
